// LSTMModel_82557861364082
// MI455X (gfx1250) — compile-verified, matrix operand order checked by reading, not run on hardware
//
#include <hip/hip_runtime.h>

typedef __attribute__((ext_vector_type(16))) _Float16 v16h;
typedef __attribute__((ext_vector_type(8)))  float    v8f;

#define T_STEPS 200
#define BATCH   4096
#define HID     64
#define IN_DIM  2

// LDS row strides (halves), padded for conflict-free ds_load_b128
#define B_ROW_H 104   // 96 data + 8 pad -> 208 bytes/row (13*16B)
#define H_ROW_H 72    // 64 data + 8 pad -> 144 bytes/row (9*16B)
#define WPW     2     // waves per workgroup

union Frag16 {
  v16h      v;
  uint4     q[2];
  uint32_t  w[8];
  _Float16  e[16];
};

__device__ __forceinline__ v8f wmma16(v16h a, v16h b, v8f c) {
  // D = A(16x32 f16) x B(32x16 f16) + C(f32)
  return __builtin_amdgcn_wmma_f32_16x16x32_f16(false, a, false, b, (short)0, c,
                                                false, false);
}

#if __has_builtin(__builtin_amdgcn_tanhf)
__device__ __forceinline__ float ftanh(float x) { return __builtin_amdgcn_tanhf(x); }
#else
__device__ __forceinline__ float ftanh(float x) {
  float e = __expf(2.0f * x);
  return 1.0f - 2.0f / (e + 1.0f);
}
#endif

// ---------------------------------------------------------------------------
// Prep: gate-interleaved f16 B-image [256 N-rows][96 K-cols] in global ws.
// Row r: u = r>>6 (unit group), g = (r>>4)&3 (gate i/f/g/o), j = r&15.
// Source W row s = g*64 + u*16 + j.
// Cols: [0..63] = W_hh[s][k]; 64 = W_ih[s][0]; 65 = W_ih[s][1];
//       66 = b_ih[s]+b_hh[s]; 67..95 = 0.
// Sigmoid gates (i,f,o) are PRE-SCALED by 0.5 so the kernel can use
// sigmoid(x) = 0.5*tanh(0.5x)+0.5 without the 0.5x multiply.
// ---------------------------------------------------------------------------
__global__ void lstm_prep(const float* __restrict__ Wih, const float* __restrict__ Whh,
                          const float* __restrict__ bih, const float* __restrict__ bhh,
                          _Float16* __restrict__ Bg) {
  int r = blockIdx.x * blockDim.x + threadIdx.x;
  if (r >= 256) return;
  int u = r >> 6, g = (r >> 4) & 3, j = r & 15;
  int s = g * 64 + u * 16 + j;
  float scale = (g == 2) ? 1.0f : 0.5f;   // cell gate keeps full scale
  _Float16* dst = Bg + r * 96;
  for (int k = 0; k < 64; ++k) dst[k] = (_Float16)(scale * Whh[s * 64 + k]);
  dst[64] = (_Float16)(scale * Wih[s * 2 + 0]);
  dst[65] = (_Float16)(scale * Wih[s * 2 + 1]);
  dst[66] = (_Float16)(scale * (bih[s] + bhh[s]));
  for (int k = 67; k < 96; ++k) dst[k] = (_Float16)0.0f;
}

// ---------------------------------------------------------------------------
// Persistent LSTM: one wave owns 16 batch rows for all 200 steps.
// No barriers inside the time loop (h buffer is per-wave, DS is in-order).
// B fragments are loop-invariant LDS reads -> LLVM promotes them to VGPRs.
// ---------------------------------------------------------------------------
__global__ __launch_bounds__(WPW * 32) void lstm_persistent(
    const float* __restrict__ x, const _Float16* __restrict__ Bg,
    const float* __restrict__ Wfc, const float* __restrict__ bfc,
    float* __restrict__ out) {
  __shared__ __align__(16) _Float16 Bsh[256 * B_ROW_H];
  __shared__ __align__(16) _Float16 Hsh[WPW * 16 * H_ROW_H];

  const int  tid  = threadIdx.x;
  const int  wv   = tid >> 5;
  const int  lane = tid & 31;
  const int  ln16 = lane & 15;
  const bool hi   = (lane & 16) != 0;

  // Cooperative copy of B image into padded LDS (3072 x 16B chunks).
  for (int idx = tid; idx < 256 * 12; idx += WPW * 32) {
    int r = idx / 12, c = idx % 12;
    uint4 v = ((const uint4*)Bg)[idx];
    *(uint4*)((char*)Bsh + r * (B_ROW_H * 2) + c * 16) = v;
  }
  __syncthreads();

  _Float16* hb = Hsh + wv * 16 * H_ROW_H;
  for (int i = lane; i < 16 * H_ROW_H; i += 32) hb[i] = (_Float16)0.0f;

  float creg[4][8];
#pragma unroll
  for (int u = 0; u < 4; ++u)
#pragma unroll
    for (int e = 0; e < 8; ++e) creg[u][e] = 0.0f;

  const int   b0   = (blockIdx.x * WPW + wv) * 16;
  const char* hrow = (const char*)hb + ln16 * (H_ROW_H * 2) + (hi ? 16 : 0);
  const char* Brow = (const char*)Bsh + ln16 * (B_ROW_H * 2) + (hi ? 16 : 0);
  const float* xbase = x + (size_t)(b0 + ln16) * T_STEPS * IN_DIM;

  // Software-pipelined x: load for step t+1 while computing step t.
  float2 xv = *(const float2*)(xbase + 0);

  for (int t = 0; t < T_STEPS; ++t) {
    const int tn = (t < T_STEPS - 1) ? (t + 1) : t;
    const float2 xnext = *(const float2*)(xbase + tn * IN_DIM);

    // --- A3 fragment: K=64..95 holds [x0, x1, 1, 0...] (lanes 0..15 only) ---
    union { _Float16 h[2]; uint32_t u32; } px;
    px.h[0] = (_Float16)xv.x;
    px.h[1] = (_Float16)xv.y;
    Frag16 a3;
    a3.w[0] = hi ? 0u : px.u32;        // K=64,65 : x0, x1
    a3.w[1] = hi ? 0u : 0x00003C00u;   // K=66    : 1.0h (bias column)
    a3.w[2] = 0u; a3.w[3] = 0u;
    a3.w[4] = 0u; a3.w[5] = 0u; a3.w[6] = 0u; a3.w[7] = 0u;

    // --- A1/A2 fragments: previous h tile (16x64 f16) from per-wave LDS ---
    Frag16 a1, a2;
    a1.q[0] = *(const uint4*)(hrow + 0);
    a1.q[1] = *(const uint4*)(hrow + 32);
    a2.q[0] = *(const uint4*)(hrow + 64);
    a2.q[1] = *(const uint4*)(hrow + 96);

    // --- Phase 1: 16 output tiles x 3 chained WMMAs, breadth-first so
    //     dependent WMMAs are 15 instructions apart (no hazard stalls). ---
    v8f acc[16];
#pragma unroll
    for (int tix = 0; tix < 16; ++tix) {
      const char* Bt = Brow + (tix * 16) * (B_ROW_H * 2);
      Frag16 f0; f0.q[0] = *(const uint4*)(Bt + 0);  f0.q[1] = *(const uint4*)(Bt + 32);
      v8f z = {};
      acc[tix] = wmma16(a1.v, f0.v, z);              // h[0:32) @ W
    }
#pragma unroll
    for (int tix = 0; tix < 16; ++tix) {
      const char* Bt = Brow + (tix * 16) * (B_ROW_H * 2);
      Frag16 f1; f1.q[0] = *(const uint4*)(Bt + 64); f1.q[1] = *(const uint4*)(Bt + 96);
      acc[tix] = wmma16(a2.v, f1.v, acc[tix]);       // h[32:64) @ W
    }
#pragma unroll
    for (int tix = 0; tix < 16; ++tix) {
      const char* Bt = Brow + (tix * 16) * (B_ROW_H * 2);
      Frag16 f2; f2.q[0] = *(const uint4*)(Bt + 128); f2.q[1] = *(const uint4*)(Bt + 160);
      acc[tix] = wmma16(a3.v, f2.v, acc[tix]);       // [x0,x1,1] @ [W_ih|bias]
    }

    // --- Phase 2: LSTM cell update (weights pre-scaled: sigmoid = 0.5*tanh+0.5)
#pragma unroll
    for (int u = 0; u < 4; ++u) {
#pragma unroll
      for (int e = 0; e < 8; ++e) {
        float ti = ftanh(acc[u * 4 + 0][e]);
        float tf = ftanh(acc[u * 4 + 1][e]);
        float tg = ftanh(acc[u * 4 + 2][e]);
        float to = ftanh(acc[u * 4 + 3][e]);
        float ig = fmaf(ti, 0.5f, 0.5f);
        float fg = fmaf(tf, 0.5f, 0.5f);
        float og = fmaf(to, 0.5f, 0.5f);
        float c  = fg * creg[u][e] + ig * tg;
        creg[u][e] = c;
        float h = og * ftanh(c);
        int m = e + (hi ? 8 : 0);
        hb[m * H_ROW_H + u * 16 + ln16] = (_Float16)h;  // row-major h for A loads
      }
    }
    xv = xnext;
  }

  // --- Epilogue: out = h_last @ W_fc^T + b_fc  (tiny: 2 outputs/row) ---
  if (lane < 16) {
    float o0 = bfc[0], o1 = bfc[1];
    const _Float16* hr = hb + lane * H_ROW_H;
#pragma unroll 8
    for (int k = 0; k < HID; ++k) {
      float hv = (float)hr[k];
      o0 += hv * Wfc[k];
      o1 += hv * Wfc[HID + k];
    }
    *(float2*)(out + (size_t)(b0 + lane) * 2) = make_float2(o0, o1);
  }
}

// ---------------------------------------------------------------------------
extern "C" void kernel_launch(void* const* d_in, const int* in_sizes, int n_in,
                              void* d_out, int out_size, void* d_ws, size_t ws_size,
                              hipStream_t stream) {
  (void)in_sizes; (void)n_in; (void)out_size; (void)ws_size;
  const float* x   = (const float*)d_in[0];
  const float* Wih = (const float*)d_in[1];
  const float* Whh = (const float*)d_in[2];
  const float* bih = (const float*)d_in[3];
  const float* bhh = (const float*)d_in[4];
  const float* Wfc = (const float*)d_in[5];
  const float* bfc = (const float*)d_in[6];
  float*    out = (float*)d_out;
  _Float16* Bg  = (_Float16*)d_ws;  // 48KB f16 weight image

  lstm_prep<<<1, 256, 0, stream>>>(Wih, Whh, bih, bhh, Bg);
  lstm_persistent<<<BATCH / (16 * WPW), WPW * 32, 0, stream>>>(x, Bg, Wfc, bfc, out);
}
